// MultiHeadAttention_90795608638325
// MI455X (gfx1250) — matrix operand order repaired and checked by reading, not compiled, not run on hardware
//
#include <hip/hip_runtime.h>

// Problem constants
constexpr int Bb = 2, Tt = 2048, Ee = 768, Hh = 12, Dd = 64; // B,T,E,H,HS

typedef __attribute__((ext_vector_type(16))) __bf16 v16bf;
typedef __attribute__((ext_vector_type(8)))  float  v8f;

union FragU { v16bf v; unsigned int u[8]; };

__device__ __forceinline__ unsigned short f2bf(float f) {
    union { float f; unsigned int u; } c; c.f = f;
    unsigned int u = c.u;
    u += 0x7FFFu + ((u >> 16) & 1u);   // round-to-nearest-even
    return (unsigned short)(u >> 16);
}

// A-fragment 16x32 bf16: lane m=lane&15 holds row m; VGPR j holds K pair at
// k = 16*(j>>2) + 8*half + 2*(j&3). K-dim contiguous in memory at rowptr.
__device__ __forceinline__ v16bf load_A32(const unsigned short* rowptr, int hf) {
    FragU f;
#pragma unroll
    for (int j = 0; j < 8; ++j) {
        int k = ((j >> 2) << 4) + (hf << 3) + ((j & 3) << 1);
        f.u[j] = *(const unsigned int*)(rowptr + k);
    }
    return f.v;
}

// B-fragment 32x16 bf16: lane n=lane&15 holds column n; VGPR j holds K pair at
// k = 16*(j>>2) + 8*half + 2*(j&3). K-dim contiguous in memory at colptr.
__device__ __forceinline__ v16bf load_B32(const unsigned short* colptr, int hf) {
    FragU f;
#pragma unroll
    for (int j = 0; j < 8; ++j)
        f.u[j] = *(const unsigned int*)(colptr + ((j >> 2) << 4) + (hf << 3) + ((j & 3) << 1));
    return f.v;
}

__device__ __forceinline__ v8f wmma_bf16(v16bf a, v16bf b, v8f c) {
    return __builtin_amdgcn_wmma_f32_16x16x32_bf16(false, a, false, b, (short)0, c, false, false);
}

// ---------------------------------------------------------------- converts
__global__ void mha_cvt_bf16(const float* __restrict__ src, unsigned short* __restrict__ dst, int n) {
    int i = blockIdx.x * 256 + threadIdx.x;
    if (i < n) dst[i] = f2bf(src[i]);
}

// [H][E][HS] fp32 -> [H][HS][E] bf16
__global__ void mha_cvt_wT(const float* __restrict__ src, unsigned short* __restrict__ dst) {
    int i = blockIdx.x * 256 + threadIdx.x;
    int n = Hh * Ee * Dd;
    if (i >= n) return;
    int h = i / (Ee * Dd);
    int r = i - h * (Ee * Dd);
    int e = r / Dd;
    int d = r - e * Dd;
    dst[(h * Dd + d) * Ee + e] = f2bf(src[i]);
}

// ---------------------------------------------------------------- QKV projection
// wave -> (mat, b, h, row-tile). Q,K: [B,H,T,HS] bf16. V stored transposed: [B,H,HS,T].
__global__ void mha_qkv(const unsigned short* __restrict__ xb,
                        const unsigned short* __restrict__ Wqt,
                        const unsigned short* __restrict__ Wkt,
                        const unsigned short* __restrict__ Wvt,
                        unsigned short* __restrict__ Q,
                        unsigned short* __restrict__ K,
                        unsigned short* __restrict__ Vt) {
    int gwave = (blockIdx.x * blockDim.x + threadIdx.x) >> 5;
    int lane  = threadIdx.x & 31;
    int n = lane & 15, hf = lane >> 4;

    int rt   = gwave & 127;        // T/16 = 128
    int rest = gwave >> 7;
    int h    = rest % Hh; rest /= Hh;
    int b    = rest & 1;
    int mat  = rest >> 1;

    const unsigned short* Wt = (mat == 0) ? Wqt : (mat == 1 ? Wkt : Wvt);
    int t0 = rt * 16;

    v8f acc[4];
#pragma unroll
    for (int t = 0; t < 4; ++t) acc[t] = (v8f){0,0,0,0,0,0,0,0};

    const unsigned short* xrow  = xb + (size_t)(b * Tt + t0 + n) * Ee;
    const unsigned short* wbase = Wt + (size_t)h * Dd * Ee;

    for (int e0 = 0; e0 < Ee; e0 += 32) {
        v16bf a = load_A32(xrow + e0, hf);
#pragma unroll
        for (int t = 0; t < 4; ++t) {
            v16bf bw = load_B32(wbase + (size_t)(t * 16 + n) * Ee + e0, hf);
            acc[t] = wmma_bf16(a, bw, acc[t]);
        }
    }

    size_t bh = (size_t)(b * Hh + h);
    if (mat == 2) {
#pragma unroll
        for (int t = 0; t < 4; ++t)
#pragma unroll
            for (int j = 0; j < 8; ++j)
                Vt[(bh * Dd + t * 16 + n) * Tt + t0 + j + hf * 8] = f2bf(acc[t][j]);
    } else {
        unsigned short* dst = (mat == 0) ? Q : K;
#pragma unroll
        for (int t = 0; t < 4; ++t)
#pragma unroll
            for (int j = 0; j < 8; ++j)
                dst[(bh * Tt + t0 + j + hf * 8) * Dd + t * 16 + n] = f2bf(acc[t][j]);
    }
}

// ---------------------------------------------------------------- flash attention
// wave -> (b, h, 16-query tile). Online softmax over 32-key blocks.
__global__ void mha_attn(const unsigned short* __restrict__ Q,
                         const unsigned short* __restrict__ K,
                         const unsigned short* __restrict__ Vt,
                         const int* __restrict__ amask,
                         unsigned short* __restrict__ attnout) {
    __shared__ unsigned short sP[4][16 * 32];
    int gwave = (blockIdx.x * blockDim.x + threadIdx.x) >> 5;
    int lane  = threadIdx.x & 31;
    int wslot = threadIdx.x >> 5;
    int n = lane & 15, hf = lane >> 4;

    int qt   = gwave & 127;
    int rest = gwave >> 7;
    int h    = rest % Hh;
    int b    = rest / Hh;
    int q0   = qt * 16;

    size_t bh = (size_t)(b * Hh + h);
    const unsigned short* Qb = Q  + bh * Tt * Dd;
    const unsigned short* Kb = K  + bh * Tt * Dd;
    const unsigned short* Vb = Vt + bh * Dd * Tt;
    unsigned short* myP = sP[wslot];

    v16bf aq0 = load_A32(Qb + (size_t)(q0 + n) * Dd,      hf);
    v16bf aq1 = load_A32(Qb + (size_t)(q0 + n) * Dd + 32, hf);

    v8f acc[4];
#pragma unroll
    for (int t = 0; t < 4; ++t) acc[t] = (v8f){0,0,0,0,0,0,0,0};
    float mi[8], li[8];
#pragma unroll
    for (int j = 0; j < 8; ++j) { mi[j] = -INFINITY; li[j] = 0.f; }

    int kbmax = (q0 + 15) >> 5;   // inclusive
    for (int kb = 0; kb <= kbmax; ++kb) {
        int k0 = kb * 32;
        int key0 = k0 + n, key1 = k0 + 16 + n;

        v8f S0 = (v8f){0,0,0,0,0,0,0,0};
        v8f S1 = (v8f){0,0,0,0,0,0,0,0};
        S0 = wmma_bf16(aq0, load_B32(Kb + (size_t)key0 * Dd,      hf), S0);
        S0 = wmma_bf16(aq1, load_B32(Kb + (size_t)key0 * Dd + 32, hf), S0);
        S1 = wmma_bf16(aq0, load_B32(Kb + (size_t)key1 * Dd,      hf), S1);
        S1 = wmma_bf16(aq1, load_B32(Kb + (size_t)key1 * Dd + 32, hf), S1);

        int am0 = amask[b * Tt + key0];
        int am1 = amask[b * Tt + key1];

        float cr[8];
#pragma unroll
        for (int j = 0; j < 8; ++j) {
            int qrow = q0 + j + hf * 8;
            float s0 = S0[j] * 0.125f;             // 1/sqrt(64)
            float s1 = S1[j] * 0.125f;
            if (key0 > qrow || am0 == 0) s0 = -1e30f;
            if (key1 > qrow || am1 == 0) s1 = -1e30f;
            float mx = fmaxf(s0, s1);
            mx = fmaxf(mx, __shfl_xor(mx, 1, 32));
            mx = fmaxf(mx, __shfl_xor(mx, 2, 32));
            mx = fmaxf(mx, __shfl_xor(mx, 4, 32));
            mx = fmaxf(mx, __shfl_xor(mx, 8, 32));
            float mnew = fmaxf(mi[j], mx);
            float corr = __expf(mi[j] - mnew);
            float p0 = __expf(s0 - mnew);
            float p1 = __expf(s1 - mnew);
            float rs = p0 + p1;
            rs += __shfl_xor(rs, 1, 32);
            rs += __shfl_xor(rs, 2, 32);
            rs += __shfl_xor(rs, 4, 32);
            rs += __shfl_xor(rs, 8, 32);
            li[j] = li[j] * corr + rs;
            mi[j] = mnew;
            cr[j] = corr;
            myP[(j + hf * 8) * 32 + n]      = f2bf(p0);
            myP[(j + hf * 8) * 32 + 16 + n] = f2bf(p1);
        }
#pragma unroll
        for (int t = 0; t < 4; ++t)
#pragma unroll
            for (int j = 0; j < 8; ++j)
                acc[t][j] *= cr[j];

        // LDS is in-order within a wave: ds loads below see the stores above.
        v16bf pa = load_A32(myP + n * 32, hf);
#pragma unroll
        for (int t = 0; t < 4; ++t) {
            v16bf bv = load_B32(Vb + (size_t)(t * 16 + n) * Tt + k0, hf);
            acc[t] = wmma_bf16(pa, bv, acc[t]);
        }
    }

#pragma unroll
    for (int t = 0; t < 4; ++t)
#pragma unroll
        for (int j = 0; j < 8; ++j) {
            float o = acc[t][j] / li[j];
            attnout[(size_t)(b * Tt + q0 + j + hf * 8) * Ee + h * Dd + t * 16 + n] = f2bf(o);
        }
}

// ---------------------------------------------------------------- output projection
// out[r, :] = attnout[r, :] @ Wo^T + bo ; B-frag K-dim is Wo's row (contiguous).
__global__ void mha_proj(const unsigned short* __restrict__ attnout,
                         const unsigned short* __restrict__ Wob,
                         const float* __restrict__ bo,
                         float* __restrict__ out) {
    int gwave = (blockIdx.x * blockDim.x + threadIdx.x) >> 5;
    int lane  = threadIdx.x & 31;
    int n = lane & 15, hf = lane >> 4;

    int g  = gwave % (Ee / 64);        // 12 column groups of 64
    int rt = gwave / (Ee / 64);        // 256 row tiles over B*T
    int t0 = rt * 16;

    v8f acc[4];
#pragma unroll
    for (int t = 0; t < 4; ++t) acc[t] = (v8f){0,0,0,0,0,0,0,0};

    const unsigned short* arow = attnout + (size_t)(t0 + n) * Ee;
    for (int e0 = 0; e0 < Ee; e0 += 32) {
        v16bf a = load_A32(arow + e0, hf);
#pragma unroll
        for (int t = 0; t < 4; ++t) {
            int col = g * 64 + t * 16 + n;
            v16bf bw = load_B32(Wob + (size_t)col * Ee + e0, hf);
            acc[t] = wmma_bf16(a, bw, acc[t]);
        }
    }
#pragma unroll
    for (int t = 0; t < 4; ++t)
#pragma unroll
        for (int j = 0; j < 8; ++j) {
            int col = g * 64 + t * 16 + n;
            out[(size_t)(t0 + j + hf * 8) * Ee + col] = acc[t][j] + bo[col];
        }
}

extern "C" void kernel_launch(void* const* d_in, const int* in_sizes, int n_in,
                              void* d_out, int out_size, void* d_ws, size_t ws_size,
                              hipStream_t stream) {
    const float* x     = (const float*)d_in[0];
    const int*   amask = (const int*)d_in[1];
    const float* Wq    = (const float*)d_in[2];
    const float* Wk    = (const float*)d_in[3];
    const float* Wv    = (const float*)d_in[4];
    const float* Wo    = (const float*)d_in[5];
    const float* bo    = (const float*)d_in[6];
    float* out = (float*)d_out;

    const size_t nX  = (size_t)Bb * Tt * Ee;     // 3,145,728
    const size_t nW  = (size_t)Hh * Ee * Dd;     //   589,824
    const size_t nWo = (size_t)Ee * Ee;          //   589,824
    const size_t nQ  = (size_t)Bb * Hh * Tt * Dd;

    unsigned short* w = (unsigned short*)d_ws;
    size_t off = 0;
    unsigned short* xb   = w + off; off += nX;
    unsigned short* Wqt  = w + off; off += nW;
    unsigned short* Wkt  = w + off; off += nW;
    unsigned short* Wvt  = w + off; off += nW;
    unsigned short* Wob  = w + off; off += nWo;
    unsigned short* Qw   = w + off; off += nQ;
    unsigned short* Kw   = w + off; off += nQ;
    unsigned short* Vtw  = w + off; off += nQ;
    unsigned short* aout = w + off; off += nX;
    (void)ws_size; (void)in_sizes; (void)n_in; (void)out_size;

    mha_cvt_bf16<<<(int)((nX + 255) / 256), 256, 0, stream>>>(x, xb, (int)nX);
    mha_cvt_wT  <<<(int)((nW + 255) / 256), 256, 0, stream>>>(Wq, Wqt);
    mha_cvt_wT  <<<(int)((nW + 255) / 256), 256, 0, stream>>>(Wk, Wkt);
    mha_cvt_wT  <<<(int)((nW + 255) / 256), 256, 0, stream>>>(Wv, Wvt);
    mha_cvt_bf16<<<(int)((nWo + 255) / 256), 256, 0, stream>>>(Wo, Wob, (int)nWo);

    // 3 mats * B * H * (T/16) = 9216 waves / 4 per block
    mha_qkv <<<2304, 128, 0, stream>>>(xb, Wqt, Wkt, Wvt, Qw, Kw, Vtw);
    // B * H * (T/16) = 3072 waves
    mha_attn<<<768, 128, 0, stream>>>(Qw, Kw, Vtw, amask, aout);
    // (B*T/16) * (E/64) = 3072 waves
    mha_proj<<<768, 128, 0, stream>>>(aout, Wob, bo, out);
}
